// Model_42004780155154
// MI455X (gfx1250) — matrix operand order repaired and checked by reading, not compiled, not run on hardware
//
#include <hip/hip_runtime.h>
#include <hip/hip_bf16.h>

// ---------------------------------------------------------------------------
// LSTM seq2seq for MI455X (gfx1250, wave32).
// - Matrix math: v_wmma_f32_16x16x32_bf16 (f32 accumulate).
// - Weights staged to LDS via Tensor Data Mover (tensor_load_to_lds),
//   double-buffered with s_wait_tensorcnt; TDM pad gives conflict-free LDS.
// - 192 serial steps -> per-step kernel chain (graph-captured by harness).
// ---------------------------------------------------------------------------

#define BATCH   256
#define SEQLEN  96
#define PREDLEN 96
#define CIN     321
#define CPAD    352      // CIN padded to multiple of 32 (16B-aligned frags)
#define DM      512
#define ND      2048     // 4*DM
#define KTOT    864      // CPAD + DM
#define NPADP   336      // CIN padded to multiple of 16 for projection N

#define BROW    40       // LDS B-panel row stride in elems: 32 + 8 pad (TDM pad)
#define BUFB    (64 * BROW * 2)   // bytes per LDS B buffer (5120)

typedef __attribute__((ext_vector_type(16))) __bf16 v16bf;
typedef __attribute__((ext_vector_type(8)))  __bf16 v8bf;
typedef __attribute__((ext_vector_type(8)))  float  v8f;
typedef __attribute__((ext_vector_type(4)))  unsigned int v4u;
typedef __attribute__((ext_vector_type(8)))  int v8i;
typedef __attribute__((ext_vector_type(4)))  int v4i;

__device__ __forceinline__ __bf16 f2bf(float f) {
    unsigned u = __float_as_uint(f);
    unsigned r = (u + 0x7FFFu + ((u >> 16) & 1u)) >> 16;
    unsigned short s = (unsigned short)r;
    return *reinterpret_cast<__bf16*>(&s);
}

__device__ __forceinline__ float sigm(float x) {
    return 1.0f / (1.0f + __expf(-x));
}

// A fragment (16x32 bf16, MxK), contiguous-k order (same as B frags):
// lane = half*16 + m holds row m; elems 0..15 = K kh..kh+15 contiguous (kh = half*16).
__device__ __forceinline__ v16bf load_a_frag(const __bf16* __restrict__ A, int lda,
                                             int m0, int k0, int lane) {
    int r  = lane & 15;
    int kh = (lane >> 4) << 4;
    const __bf16* p = A + (size_t)(m0 + r) * lda + k0 + kh;
    union { v16bf v; v8bf h[2]; } u;
    u.h[0] = *(const v8bf*)(p);
    u.h[1] = *(const v8bf*)(p + 8);
    return u.v;
}

// B fragment (32x16 bf16, KxN) direct from global W (row-major, ldw):
// lane = half*16 + n holds column n, K = half*16 + (0..15) contiguous.
__device__ __forceinline__ v16bf load_b_frag(const __bf16* __restrict__ W, int ldw,
                                             int n0, int k0, int lane) {
    int c  = lane & 15;
    int kb = (lane >> 4) << 4;
    const __bf16* p = W + (size_t)(n0 + c) * ldw + k0 + kb;
    union { v16bf v; v8bf h[2]; } u;
    u.h[0] = *(const v8bf*)(p);
    u.h[1] = *(const v8bf*)(p + 8);
    return u.v;
}

// B fragment from the LDS panel (row stride BROW elems, padded by TDM).
__device__ __forceinline__ v16bf load_b_lds(const __bf16* buf, int nn, int lane) {
    int c  = nn + (lane & 15);
    int kb = (lane >> 4) << 4;
    const __bf16* p = buf + c * BROW + kb;
    union { v16bf v; v8bf h[2]; } u;
    u.h[0] = *(const v8bf*)(p);
    u.h[1] = *(const v8bf*)(p + 8);
    return u.v;
}

// ---------------------------------------------------------------------------
// TDM: async-load an 8-row x 32-col bf16 tile of Wcat (ld = KTOT) into LDS.
// D# per ISA 8.3/8.4. Pad: after every 16 DWORDs (one 64B row) insert 4 DWORDs
// -> LDS row stride 80B (bank-conflict-free ds_load_b128 fragment reads).
// This toolchain exposes the 6-arg builtin: (g0, g1, g2, g3, g4, cpol);
// groups 2/3/4 are unused for a 2-D tile -> zeros.
// ---------------------------------------------------------------------------
__device__ __forceinline__ void tdm_load_b_slab(const __bf16* gptr, unsigned lds_off) {
    unsigned long long ga = (unsigned long long)(uintptr_t)gptr;
    v4u g0;
    g0.x = 1u;                                            // count=1, user mode
    g0.y = lds_off;                                       // lds_addr (bytes)
    g0.z = (unsigned)(ga & 0xFFFFFFFFu);                  // global_addr[31:0]
    g0.w = (unsigned)((ga >> 32) & 0x01FFFFFFu) | (2u << 30); // [56:32] | type=2
    v8i g1;
    g1[0] = 0x06D10000;                  // data_size=2B | pad_en | intv=16dw | amt=4dw
    g1[1] = (int)(((unsigned)KTOT) << 16);        // tensor_dim0[15:0] in [31:16]
    g1[2] = (int)(((unsigned)ND) << 16);          // dim0 hi=0 | tensor_dim1[15:0]
    g1[3] = (int)(32u << 16);                     // dim1 hi=0 | tile_dim0=32
    g1[4] = 8;                                    // tile_dim1=8 | tile_dim2=0
    g1[5] = KTOT;                                 // tensor_dim0_stride lo
    g1[6] = 0;                                    // stride0 hi | stride1 lo
    g1[7] = 0;                                    // stride1 hi
    v4i z4 = {0, 0, 0, 0};                        // 2-D tile: groups 2/3 unused
    v8i z8 = {0, 0, 0, 0, 0, 0, 0, 0};
    __builtin_amdgcn_tensor_load_to_lds(g0, g1, z4, z4, z8, 0);
}

// ---------------------------------------------------------------------------
// One-time preprocessing kernels
// ---------------------------------------------------------------------------

__global__ void norm_stats_kernel(const float* __restrict__ x,
                                  float* __restrict__ means,
                                  float* __restrict__ stdev) {
    int idx = blockIdx.x * blockDim.x + threadIdx.x;
    if (idx >= BATCH * CIN) return;
    int b = idx / CIN, c = idx % CIN;
    float s = 0.f, sq = 0.f;
    const float* p = x + (size_t)b * SEQLEN * CIN + c;
    #pragma unroll 4
    for (int t = 0; t < SEQLEN; ++t) {
        float v = p[(size_t)t * CIN];
        s += v; sq += v * v;
    }
    float mean = s * (1.0f / SEQLEN);
    float var  = sq * (1.0f / SEQLEN) - mean * mean;
    means[idx] = mean;
    stdev[idx] = sqrtf(var + 1e-5f);
}

// xn layout: [t][b][CPAD] bf16, pad cols zeroed.
__global__ void norm_apply_kernel(const float* __restrict__ x,
                                  const float* __restrict__ means,
                                  const float* __restrict__ stdev,
                                  __bf16* __restrict__ xn) {
    int idx = blockIdx.x * blockDim.x + threadIdx.x;
    if (idx >= SEQLEN * BATCH * CPAD) return;
    int c = idx % CPAD;
    int b = (idx / CPAD) % BATCH;
    int t = idx / (CPAD * BATCH);
    float v = 0.f;
    if (c < CIN) {
        int bc = b * CIN + c;
        v = (x[((size_t)b * SEQLEN + t) * CIN + c] - means[bc]) / stdev[bc];
    }
    xn[idx] = f2bf(v);
}

// Wcat[n][k] (2048 x 864): k<321 -> Wih[n][k]; 321<=k<352 -> 0; else Whh[n][k-352]
__global__ void convert_wcat_kernel(const float* __restrict__ Wih,
                                    const float* __restrict__ Whh,
                                    __bf16* __restrict__ Wcat) {
    int idx = blockIdx.x * blockDim.x + threadIdx.x;
    if (idx >= ND * KTOT) return;
    int n = idx / KTOT, k = idx % KTOT;
    float v = 0.f;
    if (k < CIN)        v = Wih[(size_t)n * CIN + k];
    else if (k >= CPAD) v = Whh[(size_t)n * DM + (k - CPAD)];
    Wcat[idx] = f2bf(v);
}

// Wpb[n][k] (336 x 512), rows >= 321 zero.
__global__ void convert_wp_kernel(const float* __restrict__ Wp,
                                  __bf16* __restrict__ Wpb) {
    int idx = blockIdx.x * blockDim.x + threadIdx.x;
    if (idx >= NPADP * DM) return;
    int n = idx / DM, k = idx % DM;
    float v = (n < CIN) ? Wp[(size_t)n * DM + k] : 0.f;
    Wpb[idx] = f2bf(v);
}

__global__ void init_state_kernel(__bf16* __restrict__ Hbuf,
                                  float* __restrict__ cbuf,
                                  __bf16* __restrict__ ybuf) {
    int idx = blockIdx.x * blockDim.x + threadIdx.x;
    __bf16 z = f2bf(0.f);
    if (idx < BATCH * DM) { Hbuf[idx] = z; cbuf[idx] = 0.f; }
    if (idx < BATCH * CPAD) ybuf[idx] = z;   // pad cols must stay zero
}

// ---------------------------------------------------------------------------
// Gates GEMM: G(256x2048) = [Xpart(256xCPAD) | Hbuf(256x512)] x Wcat^T
// Block = 8 waves; block tile 128M x 64N; wave tile 16M x 64N.
// B panel staged to LDS by TDM, double-buffered on TENSORcnt; each wave DMAs
// its own 8-row slab (no divergent guard around EXEC-ignoring TDM ops).
// Grid = (ND/64, BATCH/128) = (32, 2).
// ---------------------------------------------------------------------------
__global__ __launch_bounds__(256)
void gemm_gates_kernel(const __bf16* __restrict__ X,    // stride CPAD
                       const __bf16* __restrict__ H,    // stride DM
                       const __bf16* __restrict__ W,    // stride KTOT
                       float* __restrict__ G) {         // stride ND
    __shared__ __align__(16) __bf16 Bsm[2][64 * BROW];

    int lane = threadIdx.x & 31;
    int wave = threadIdx.x >> 5;
    int m0 = blockIdx.y * 128 + wave * 16;
    int n0 = blockIdx.x * 64;

    unsigned lds_base = (unsigned)(uintptr_t)(&Bsm[0][0]);
    unsigned lds_wave = lds_base + (unsigned)(wave * 8 * BROW * 2);
    const __bf16* wrow = W + (size_t)(n0 + wave * 8) * KTOT;

    const int NT = KTOT / 32;            // 27 k-slabs
    tdm_load_b_slab(wrow, lds_wave);     // prefetch slab 0

    v8f a0 = {}, a1 = {}, a2 = {}, a3 = {};
    for (int kt = 0; kt < NT; ++kt) {
        int k0 = kt * 32;
        if (kt + 1 < NT) {
            tdm_load_b_slab(wrow + k0 + 32, lds_wave + (unsigned)(((kt + 1) & 1) * BUFB));
            __builtin_amdgcn_s_wait_tensorcnt(1);   // slab kt complete (in-order)
        } else {
            __builtin_amdgcn_s_wait_tensorcnt(0);
        }
        __syncthreads();                            // panel visible to all waves

        v16bf a = (k0 < CPAD) ? load_a_frag(X, CPAD, m0, k0, lane)
                              : load_a_frag(H, DM,   m0, k0 - CPAD, lane);
        const __bf16* bufp = &Bsm[kt & 1][0];
        v16bf b0 = load_b_lds(bufp,  0, lane);
        v16bf b1 = load_b_lds(bufp, 16, lane);
        v16bf b2 = load_b_lds(bufp, 32, lane);
        v16bf b3 = load_b_lds(bufp, 48, lane);
        a0 = __builtin_amdgcn_wmma_f32_16x16x32_bf16(false, a, false, b0, (short)0, a0, false, false);
        a1 = __builtin_amdgcn_wmma_f32_16x16x32_bf16(false, a, false, b1, (short)0, a1, false, false);
        a2 = __builtin_amdgcn_wmma_f32_16x16x32_bf16(false, a, false, b2, (short)0, a2, false, false);
        a3 = __builtin_amdgcn_wmma_f32_16x16x32_bf16(false, a, false, b3, (short)0, a3, false, false);

        __syncthreads();                            // all reads done before overwrite
    }

    // C/D layout: lane = half*16 + n; rows m0 + half*8 + r, r = vgpr index.
    int col = (lane & 15);
    int row = m0 + ((lane >> 4) << 3);
    float* Cp = G + (size_t)row * ND + n0 + col;
    #pragma unroll
    for (int r = 0; r < 8; ++r) {
        Cp[(size_t)r * ND +  0] = a0[r];
        Cp[(size_t)r * ND + 16] = a1[r];
        Cp[(size_t)r * ND + 32] = a2[r];
        Cp[(size_t)r * ND + 48] = a3[r];
    }
}

// ---------------------------------------------------------------------------
// Fused LSTM gate activations + state update. h written as bf16 (next A).
// ---------------------------------------------------------------------------
__global__ void lstm_update_kernel(const float* __restrict__ G,
                                   const float* __restrict__ bih,
                                   const float* __restrict__ bhh,
                                   float* __restrict__ cbuf,
                                   __bf16* __restrict__ Hbuf) {
    int idx = blockIdx.x * blockDim.x + threadIdx.x;
    if (idx >= BATCH * DM) return;
    int b = idx >> 9;          // /DM
    int d = idx & (DM - 1);
    const float* g = G + (size_t)b * ND;
    float gi = g[d]          + bih[d]          + bhh[d];
    float gf = g[DM + d]     + bih[DM + d]     + bhh[DM + d];
    float gg = g[2 * DM + d] + bih[2 * DM + d] + bhh[2 * DM + d];
    float go = g[3 * DM + d] + bih[3 * DM + d] + bhh[3 * DM + d];
    float c  = sigm(gf) * cbuf[idx] + sigm(gi) * tanhf(gg);
    float h  = sigm(go) * tanhf(c);
    cbuf[idx] = c;
    Hbuf[idx] = f2bf(h);
}

// ---------------------------------------------------------------------------
// Decoder projection: y = h @ Wp^T + bp, fused de-norm store to out[:, t, :]
// and bf16 store to ybuf (next GEMM's X). One 16x16 tile per wave.
// Tiles: 16 (M) x 21 (N=336/16) = 336 waves; grid = 42 blocks x 8 waves.
// ---------------------------------------------------------------------------
__global__ __launch_bounds__(256)
void proj_kernel(const __bf16* __restrict__ H,     // 256 x 512 bf16
                 const __bf16* __restrict__ Wpb,   // 336 x 512 bf16
                 const float* __restrict__ bp,
                 const float* __restrict__ means,
                 const float* __restrict__ stdev,
                 float* __restrict__ out,
                 __bf16* __restrict__ ybuf,
                 int t) {
    int lane = threadIdx.x & 31;
    int wave = threadIdx.x >> 5;
    int tile = blockIdx.x * 8 + wave;
    if (tile >= 16 * (NPADP / 16)) return;
    int m0 = (tile & 15) * 16;
    int n0 = (tile >> 4) * 16;

    v8f acc = {};
    for (int k0 = 0; k0 < DM; k0 += 32) {
        v16bf a = load_a_frag(H, DM, m0, k0, lane);
        v16bf b = load_b_frag(Wpb, DM, n0, k0, lane);
        acc = __builtin_amdgcn_wmma_f32_16x16x32_bf16(false, a, false, b, (short)0, acc, false, false);
    }

    int col = n0 + (lane & 15);
    if (col >= CIN) return;
    int row = m0 + ((lane >> 4) << 3);
    float bpc = bp[col];
    #pragma unroll
    for (int r = 0; r < 8; ++r) {
        int b_ = row + r;
        float y = acc[r] + bpc;
        int bc = b_ * CIN + col;
        out[((size_t)b_ * PREDLEN + t) * CIN + col] = y * stdev[bc] + means[bc];
        ybuf[(size_t)b_ * CPAD + col] = f2bf(y);
    }
}

// ---------------------------------------------------------------------------
// Host launcher
// ---------------------------------------------------------------------------
static inline size_t al256(size_t x) { return (x + 255) & ~(size_t)255; }

extern "C" void kernel_launch(void* const* d_in, const int* in_sizes, int n_in,
                              void* d_out, int out_size, void* d_ws, size_t ws_size,
                              hipStream_t stream) {
    const float* x_enc = (const float*)d_in[0];
    const float* Wih   = (const float*)d_in[1];
    const float* Whh   = (const float*)d_in[2];
    const float* bih   = (const float*)d_in[3];
    const float* bhh   = (const float*)d_in[4];
    const float* Wp    = (const float*)d_in[5];
    const float* bp    = (const float*)d_in[6];
    float* out = (float*)d_out;

    char* ws = (char*)d_ws;
    size_t off = 0;
    float*  means = (float*)(ws + off);  off = al256(off + sizeof(float) * BATCH * CIN);
    float*  stdev = (float*)(ws + off);  off = al256(off + sizeof(float) * BATCH * CIN);
    __bf16* xn    = (__bf16*)(ws + off); off = al256(off + sizeof(__bf16) * (size_t)SEQLEN * BATCH * CPAD);
    __bf16* Wcat  = (__bf16*)(ws + off); off = al256(off + sizeof(__bf16) * (size_t)ND * KTOT);
    __bf16* Wpb   = (__bf16*)(ws + off); off = al256(off + sizeof(__bf16) * (size_t)NPADP * DM);
    float*  gates = (float*)(ws + off);  off = al256(off + sizeof(float) * (size_t)BATCH * ND);
    __bf16* Hbuf  = (__bf16*)(ws + off); off = al256(off + sizeof(__bf16) * (size_t)BATCH * DM);
    float*  cbuf  = (float*)(ws + off);  off = al256(off + sizeof(float) * (size_t)BATCH * DM);
    __bf16* ybuf  = (__bf16*)(ws + off); off = al256(off + sizeof(__bf16) * (size_t)BATCH * CPAD);

    // --- preprocessing ---
    {
        int n = BATCH * CIN;
        norm_stats_kernel<<<(n + 255) / 256, 256, 0, stream>>>(x_enc, means, stdev);
    }
    {
        int n = SEQLEN * BATCH * CPAD;
        norm_apply_kernel<<<(n + 255) / 256, 256, 0, stream>>>(x_enc, means, stdev, xn);
    }
    {
        int n = ND * KTOT;
        convert_wcat_kernel<<<(n + 255) / 256, 256, 0, stream>>>(Wih, Whh, Wcat);
    }
    {
        int n = NPADP * DM;
        convert_wp_kernel<<<(n + 255) / 256, 256, 0, stream>>>(Wp, Wpb);
    }
    {
        int n = BATCH * DM;
        init_state_kernel<<<(n + 255) / 256, 256, 0, stream>>>(Hbuf, cbuf, ybuf);
    }

    dim3 ggrid(ND / 64, BATCH / 128, 1);      // (32, 2)
    int nupd = BATCH * DM;                    // 131072

    // --- encoder: 96 steps ---
    for (int t = 0; t < SEQLEN; ++t) {
        const __bf16* Xt = xn + (size_t)t * BATCH * CPAD;
        gemm_gates_kernel<<<ggrid, 256, 0, stream>>>(Xt, Hbuf, Wcat, gates);
        lstm_update_kernel<<<(nupd + 255) / 256, 256, 0, stream>>>(gates, bih, bhh, cbuf, Hbuf);
    }

    // --- decoder: 96 steps ---
    for (int t = 0; t < PREDLEN; ++t) {
        proj_kernel<<<42, 256, 0, stream>>>(Hbuf, Wpb, bp, means, stdev, out, ybuf, t);
        gemm_gates_kernel<<<ggrid, 256, 0, stream>>>(ybuf, Hbuf, Wcat, gates);
        lstm_update_kernel<<<(nupd + 255) / 256, 256, 0, stream>>>(gates, bih, bhh, cbuf, Hbuf);
    }
}
